// cross_Attention_39204461478436
// MI455X (gfx1250) — matrix operand order repaired and checked by reading, not compiled, not run on hardware
//
#include <hip/hip_runtime.h>
#include <hip/hip_bf16.h>

// ---------------------------------------------------------------------------
// Problem constants (from reference): B=8, C=192, H=W=128, HEADS=8
// ---------------------------------------------------------------------------
constexpr int BB   = 8;
constexpr int CC   = 192;
constexpr int HEI  = 128;
constexpr int WID  = 128;
constexpr int NPIX = HEI * WID;      // 16384
constexpr int NH   = 8;              // heads
constexpr int CH   = CC / NH;        // 24 channels per head
constexpr int KSPLIT = 8;            // K-split for the Gram stage
constexpr int KCH    = NPIX / KSPLIT;

// ---------------------------------------------------------------------------
// Workspace layout (bytes). Peak use ~193 MiB.
// ---------------------------------------------------------------------------
constexpr size_t OFF_WQKV  = 0;                          // 576*192 f16
constexpr size_t OFF_WQDW  = OFF_WQKV + 576 * 192 * 2;   // 192*192 f16
constexpr size_t OFF_WPROJ = OFF_WQDW + 192 * 192 * 2;   // 192*192 f16
constexpr size_t OFF_RQ    = OFF_WPROJ + 192 * 192 * 2;  // 8*192 f32
constexpr size_t OFF_RK    = OFF_RQ + 1536 * 4;          // 8*192 f32
constexpr size_t OFF_ATTN  = OFF_RK + 1536 * 4;          // 64*32*32 f32 (zero padded)
constexpr size_t OFF_GBUF  = OFF_ATTN + 64 * 32 * 32 * 4;// 64*32*32 f32 partial Gram
constexpr size_t OFF_SEG0  = 1u << 20;                   // 2 x 50.3MB slots (or one 384-ch buf)
constexpr size_t SEGBYTES  = (size_t)BB * CC * NPIX * 2; // 50,331,648 B ([8,192,16384] f16)
constexpr size_t OFF_SEG1  = OFF_SEG0 + 2 * SEGBYTES;    // [8,384,16384] f16 (k|v)

// ---------------------------------------------------------------------------
// WMMA helper (CDNA5 wave32, D = A(16x32 f16) * B(32x16 f16) + C(16x16 f32))
// ---------------------------------------------------------------------------
typedef __attribute__((ext_vector_type(16))) _Float16 v16h;
typedef __attribute__((ext_vector_type(8)))  float    v8f;

__device__ __forceinline__ v8f wmma_f16(v16h a, v16h b, v8f c) {
  // args: (neg_a, A, neg_b, B, c_mod, C, reuse_a, reuse_b)
  return __builtin_amdgcn_wmma_f32_16x16x32_f16(false, a, false, b, (short)0, c,
                                                false, false);
}

// ---------------------------------------------------------------------------
// fp32 -> f16 weight conversion / fp32 zero fill
// ---------------------------------------------------------------------------
__global__ void __launch_bounds__(256)
cvt_f32_to_f16(const float* __restrict__ in, _Float16* __restrict__ out, int n) {
  int i = blockIdx.x * 256 + threadIdx.x;
  if (i < n) out[i] = (_Float16)in[i];
}

__global__ void __launch_bounds__(256)
zero_f32(float* __restrict__ p, int n) {
  int i = blockIdx.x * 256 + threadIdx.x;
  if (i < n) p[i] = 0.f;
}

// ---------------------------------------------------------------------------
// 1x1 conv as GEMM: Out[b,o,n] = sum_c W[o,c] * Act[b,c,n] + bias[o]
//  - block = 128 threads (4 waves); block tile = 64 out-ch x 64 pixels
//  - each wave owns 16 pixels and 4 M-tiles (4 accumulators) so one LDS
//    B-fragment feeds 4 independent WMMAs; K stepped by 32 through LDS
//  - fragment layouts per ISA 7.12.2 tables
// ---------------------------------------------------------------------------
template <typename TIN, typename TOUT>
__global__ void __launch_bounds__(128)
conv1x1_gemm(const TIN* __restrict__ act, const _Float16* __restrict__ w16,
             const float* __restrict__ bias, TOUT* __restrict__ out,
             int Cin, int Cout) {
  __shared__ _Float16 lb[64][34];     // [pixel][k] transposed act tile (padded)
  const int tid  = threadIdx.x;
  const int lane = tid & 31;
  const int wave = tid >> 5;
  const int n0 = blockIdx.x * 64;
  const int ob = blockIdx.y * 64;     // 64 out-channels per block
  const int b  = blockIdx.z;
  const TIN* actb = act + (size_t)b * Cin * NPIX;

  const int cl  = tid >> 2;           // 0..31 : k row within tile
  const int grp = (tid & 3) * 16;     // 0,16,32,48 : pixel sub-group

  const int khalf = (lane & 16) ? 8 : 0;   // A: lanes16-31 hold K=8..15 / 24..31
  const int kb16  = (lane & 16) ? 16 : 0;  // unused; B also uses khalf (lanes16-31 hold K=8..15 / 24..31)
  const int nl    = wave * 16 + (lane & 15);

  v8f acc[4] = {};
  for (int k0 = 0; k0 < Cin; k0 += 32) {
    __syncthreads();
    // ---- cooperative, coalesced load of Act[k0..k0+32, n0..n0+64] -> LDS ----
    const TIN* src = actb + (size_t)(k0 + cl) * NPIX + n0 + grp;
    _Float16 v[16];
    if constexpr (sizeof(TIN) == 4) {           // fp32 input
      const float4* s4 = (const float4*)src;
      #pragma unroll
      for (int q = 0; q < 4; ++q) {
        float4 f = s4[q];
        v[q * 4 + 0] = (_Float16)f.x;
        v[q * 4 + 1] = (_Float16)f.y;
        v[q * 4 + 2] = (_Float16)f.z;
        v[q * 4 + 3] = (_Float16)f.w;
      }
    } else {                                    // f16 input: one 32B load
      v16h t = *(const v16h*)src;
      #pragma unroll
      for (int q = 0; q < 16; ++q) v[q] = t[q];
    }
    #pragma unroll
    for (int q = 0; q < 16; ++q) lb[grp + q][cl] = v[q];
    if (k0 + 32 < Cin)  // gfx1250 global_prefetch of next K tile
      __builtin_prefetch(actb + (size_t)(k0 + 32 + cl) * NPIX + n0 + grp, 0, 1);
    __syncthreads();

    // ---- B fragment from LDS (column = pixel, K contiguous), shared by 4 M ----
    v16h bf;
    #pragma unroll
    for (int j = 0; j < 8; ++j) { bf[j] = lb[nl][khalf + j]; bf[8 + j] = lb[nl][khalf + 16 + j]; }

    // ---- 4 M-tiles: A fragments from f16 weights (row-major, K contiguous) ----
    #pragma unroll
    for (int mi = 0; mi < 4; ++mi) {
      const int mrow = ob + mi * 16 + (lane & 15);
      const _Float16* wr = w16 + (size_t)mrow * Cin + k0 + khalf;
      v16h a;
      #pragma unroll
      for (int j = 0; j < 8; ++j) { a[j] = wr[j]; a[8 + j] = wr[16 + j]; }
      acc[mi] = wmma_f16(a, bf, acc[mi]);
    }
  }

  // ---- store D + bias ----
  const int pix  = n0 + wave * 16 + (lane & 15);
  const int mofs = (lane & 16) ? 8 : 0;
  TOUT* outb = out + (size_t)b * Cout * NPIX;
  #pragma unroll
  for (int mi = 0; mi < 4; ++mi) {
    #pragma unroll
    for (int r = 0; r < 8; ++r) {
      const int m = mi * 16 + r + mofs;
      float vv = acc[mi][r] + bias[ob + m];
      outb[(size_t)(ob + m) * NPIX + pix] = (TOUT)vv;
    }
  }
}

// ---------------------------------------------------------------------------
// Depthwise 3x3, SAME zero padding, + bias. f16 in/out, fp32 math.
// ---------------------------------------------------------------------------
__global__ void __launch_bounds__(256)
dwconv3x3(const _Float16* __restrict__ in, const float* __restrict__ w,
          const float* __restrict__ bias, _Float16* __restrict__ out,
          int Ch, long long total) {
  long long idx = (long long)blockIdx.x * 256 + threadIdx.x;
  if (idx >= total) return;
  int n  = (int)(idx % NPIX);
  int bc = (int)(idx / NPIX);
  int c  = bc % Ch;
  int px = n & (WID - 1);
  int py = n >> 7;                     // W == 128
  float acc = bias[c];
  const _Float16* inc = in + (size_t)bc * NPIX;
  const float*    wc  = w + c * 9;
  #pragma unroll
  for (int dy = -1; dy <= 1; ++dy) {
    int yy = py + dy;
    if (yy < 0 || yy >= HEI) continue;
    #pragma unroll
    for (int dx = -1; dx <= 1; ++dx) {
      int xx = px + dx;
      if (xx < 0 || xx >= WID) continue;
      acc += (float)inc[yy * WID + xx] * wc[(dy + 1) * 3 + (dx + 1)];
    }
  }
  out[idx] = (_Float16)acc;
}

// ---------------------------------------------------------------------------
// 1/max(||row||_2, 1e-12) over N=16384, fp32 accumulation.
// One block per (b, c<192); chStride lets k live inside the 384-ch k|v buffer.
// ---------------------------------------------------------------------------
__global__ void __launch_bounds__(256)
rownorm_inv(const _Float16* __restrict__ in, float* __restrict__ rinv,
            int chStride) {
  const int row = blockIdx.x;
  const int b = row / CC, c = row % CC;
  const _Float16* p = in + ((size_t)b * chStride + c) * NPIX;
  float s = 0.f;
  for (int i = threadIdx.x; i < NPIX; i += 256) {
    float v = (float)p[i];
    s += v * v;
  }
  __shared__ float red[256];
  red[threadIdx.x] = s;
  __syncthreads();
  for (int off = 128; off > 0; off >>= 1) {
    if (threadIdx.x < off) red[threadIdx.x] += red[threadIdx.x + off];
    __syncthreads();
  }
  if (threadIdx.x == 0) rinv[row] = 1.f / fmaxf(sqrtf(red[0]), 1e-12f);
}

// ---------------------------------------------------------------------------
// Gram partials: G[c,d] += sum_{n in K-chunk} q[c,n]*k[d,n] via WMMA.
// grid = (64 bh, KSPLIT); address-clamped rows (garbage lands only in rows
// >=24, never read) so the K-loop has no predication at all.
// Partials combined with global fp32 atomics into Gbuf.
// ---------------------------------------------------------------------------
__global__ void __launch_bounds__(128)
chan_gram(const _Float16* __restrict__ qy, const _Float16* __restrict__ kv,
          float* __restrict__ Gbuf) {
  const int bh = blockIdx.x;
  const int ks = blockIdx.y;
  const int b = bh >> 3, h = bh & 7;
  const int lane = threadIdx.x & 31, wave = threadIdx.x >> 5;
  const int Mi = wave >> 1, Ni = wave & 1;

  const _Float16* qb = qy + ((size_t)b * CC + h * CH) * NPIX;
  const _Float16* kb = kv + ((size_t)b * (2 * CC) + h * CH) * NPIX;  // k block

  const int cA = Mi * 16 + (lane & 15);
  const int dB = Ni * 16 + (lane & 15);
  const int cAc = (cA < CH) ? cA : (CH - 1);   // clamp: no EXEC masking
  const int dBc = (dB < CH) ? dB : (CH - 1);
  const _Float16* pa = qb + (size_t)cAc * NPIX;
  const _Float16* pb = kb + (size_t)dBc * NPIX;
  const int khalf = (lane & 16) ? 8 : 0;
  const int kb16  = (lane & 16) ? 16 : 0;

  v8f acc = {};
  const int kend = ks * KCH + KCH;
  for (int k0 = ks * KCH; k0 < kend; k0 += 32) {
    v16h a, bf;
    const _Float16* sa = pa + k0 + khalf;
    #pragma unroll
    for (int j = 0; j < 8; ++j) { a[j] = sa[j]; a[8 + j] = sa[16 + j]; }
    const _Float16* sb = pb + k0 + khalf;
    #pragma unroll
    for (int j = 0; j < 8; ++j) { bf[j] = sb[j]; bf[8 + j] = sb[16 + j]; }
    acc = wmma_f16(a, bf, acc);
  }

  const int mofs = (lane & 16) ? 8 : 0;
  float* g = Gbuf + (size_t)bh * 32 * 32;
  #pragma unroll
  for (int r = 0; r < 8; ++r) {
    const int m = Mi * 16 + r + mofs;
    atomicAdd(&g[m * 32 + Ni * 16 + (lane & 15)], acc[r]);
  }
}

// ---------------------------------------------------------------------------
// attn = softmax_d( G[c,d] * rq[c] * rk[d] * T[h] ), zero-padded to 32x32.
// ---------------------------------------------------------------------------
__global__ void __launch_bounds__(32)
attn_softmax(const float* __restrict__ Gbuf, const float* __restrict__ rq,
             const float* __restrict__ rk, const float* __restrict__ temp,
             float* __restrict__ attn) {
  const int bh = blockIdx.x;
  const int b = bh >> 3, h = bh & 7;
  const int c = threadIdx.x;
  const float* g = Gbuf + (size_t)bh * 32 * 32 + c * 32;
  float* arow = attn + ((size_t)bh * 32 + c) * 32;
  if (c < CH) {
    const float rc = rq[b * CC + h * CH + c] * temp[h];
    float lg[CH];
    float mx = -1e30f;
    #pragma unroll
    for (int d = 0; d < CH; ++d) {
      float v = g[d] * rc * rk[b * CC + h * CH + d];
      lg[d] = v;
      mx = fmaxf(mx, v);
    }
    float sum = 0.f;
    #pragma unroll
    for (int d = 0; d < CH; ++d) { lg[d] = __expf(lg[d] - mx); sum += lg[d]; }
    const float inv = 1.f / sum;
    #pragma unroll
    for (int d = 0; d < CH; ++d) arow[d] = lg[d] * inv;
    #pragma unroll
    for (int d = CH; d < 32; ++d) arow[d] = 0.f;
  } else {
    #pragma unroll
    for (int d = 0; d < 32; ++d) arow[d] = 0.f;
  }
}

// ---------------------------------------------------------------------------
// out[b, h*24+c, n] = sum_d attn[b,h,c,d] * v[b, h*24+d, n]
// K padded to 32; attn cols >=24 are zero, so B rows >=24 are address-clamped
// (their contribution is multiplied by zero) -- no predication in the loads.
// ---------------------------------------------------------------------------
__global__ void __launch_bounds__(128)
apply_attn(const float* __restrict__ attn, const _Float16* __restrict__ kv,
           _Float16* __restrict__ out) {
  const int b = blockIdx.z, h = blockIdx.y;
  const int n0 = blockIdx.x * 64;
  const int lane = threadIdx.x & 31, wave = threadIdx.x >> 5;
  const int pix = n0 + wave * 16 + (lane & 15);

  const _Float16* vbase = kv + ((size_t)b * (2 * CC) + CC + h * CH) * NPIX;
  // B fragment (shared across both M tiles); clamp d>=24 (zero A column).
  v16h bf;
  {
    const int kb16 = (lane & 16) ? 16 : 0;
    #pragma unroll
    for (int j = 0; j < 16; ++j) {
      const int d = (j < 8) ? (kb16 / 2 + j) : (16 + kb16 / 2 + (j - 8));
      const int dc = (d < CH) ? d : (CH - 1);
      bf[j] = vbase[(size_t)dc * NPIX + pix];
    }
  }
  const float* ab = attn + (size_t)(b * NH + h) * 32 * 32;
  _Float16* ob = out + ((size_t)b * CC + h * CH) * NPIX;
  const int khalf = (lane & 16) ? 8 : 0;
  const int mofs  = (lane & 16) ? 8 : 0;

  #pragma unroll
  for (int Mi = 0; Mi < 2; ++Mi) {
    const int c = Mi * 16 + (lane & 15);
    const float* ar = ab + (size_t)c * 32;     // zero-padded rows/cols
    v16h a;
    #pragma unroll
    for (int j = 0; j < 8; ++j) {
      a[j]     = (_Float16)ar[khalf + j];
      a[8 + j] = (_Float16)ar[khalf + 16 + j];
    }
    v8f acc = {};
    acc = wmma_f16(a, bf, acc);
    #pragma unroll
    for (int r = 0; r < 8; ++r) {
      const int m = Mi * 16 + r + mofs;
      if (m < CH) ob[(size_t)m * NPIX + pix] = (_Float16)acc[r];
    }
  }
}

// ---------------------------------------------------------------------------
// Host-side launch sequence (graph-capture safe; everything on `stream`).
// ---------------------------------------------------------------------------
extern "C" void kernel_launch(void* const* d_in, const int* in_sizes, int n_in,
                              void* d_out, int out_size, void* d_ws,
                              size_t ws_size, hipStream_t stream) {
  const float* x      = (const float*)d_in[0];
  const float* y      = (const float*)d_in[1];
  const float* qkv_w  = (const float*)d_in[2];   // [576,192]
  const float* qkv_b  = (const float*)d_in[3];   // [576]
  const float* dw_w   = (const float*)d_in[4];   // [576,1,3,3]
  const float* dw_b   = (const float*)d_in[5];   // [576]
  const float* qdw_w  = (const float*)d_in[6];   // [192,192]
  const float* qdw_b  = (const float*)d_in[7];   // [192]
  const float* proj_w = (const float*)d_in[8];   // [192,192]
  const float* proj_b = (const float*)d_in[9];   // [192]
  const float* temp   = (const float*)d_in[10];  // [8]

  char* ws = (char*)d_ws;
  _Float16* w_qkv16  = (_Float16*)(ws + OFF_WQKV);
  _Float16* w_qdw16  = (_Float16*)(ws + OFF_WQDW);
  _Float16* w_proj16 = (_Float16*)(ws + OFF_WPROJ);
  float*    rq       = (float*)(ws + OFF_RQ);
  float*    rk       = (float*)(ws + OFF_RK);
  float*    attnbuf  = (float*)(ws + OFF_ATTN);
  float*    gbuf     = (float*)(ws + OFF_GBUF);
  _Float16* slotA    = (_Float16*)(ws + OFF_SEG0);             // 192-ch buf
  _Float16* slotB    = (_Float16*)(ws + OFF_SEG0 + SEGBYTES);  // 192-ch buf
  _Float16* t0       = slotA;                                  // 384-ch (A+B)
  _Float16* t_kv     = (_Float16*)(ws + OFF_SEG1);             // 384-ch k|v

  // 0) weights -> f16; zero the Gram accumulator
  cvt_f32_to_f16<<<(576 * 192 + 255) / 256, 256, 0, stream>>>(qkv_w, w_qkv16,
                                                              576 * 192);
  cvt_f32_to_f16<<<(192 * 192 + 255) / 256, 256, 0, stream>>>(qdw_w, w_qdw16,
                                                              192 * 192);
  cvt_f32_to_f16<<<(192 * 192 + 255) / 256, 256, 0, stream>>>(proj_w, w_proj16,
                                                              192 * 192);
  zero_f32<<<(64 * 32 * 32 + 255) / 256, 256, 0, stream>>>(gbuf, 64 * 32 * 32);

  // 1) x branch: only k,v rows of qkv (q of x is discarded) -> t0 [8,384,N]
  conv1x1_gemm<float, _Float16><<<dim3(NPIX / 64, 384 / 64, BB), 128, 0,
                                  stream>>>(x, w_qkv16 + (size_t)CC * CC,
                                            qkv_b + CC, t0, CC, 2 * CC);
  // 2) depthwise 3x3 on k|v -> t_kv
  {
    long long tot = (long long)BB * 2 * CC * NPIX;
    dwconv3x3<<<(unsigned)((tot + 255) / 256), 256, 0, stream>>>(
        t0, dw_w + (size_t)CC * 9, dw_b + CC, t_kv, 2 * CC, tot);
  }
  // 3) y branch: qkv[:C] 1x1 -> slotA
  conv1x1_gemm<float, _Float16><<<dim3(NPIX / 64, CC / 64, BB), 128, 0,
                                  stream>>>(y, w_qkv16, qkv_b, slotA, CC, CC);
  // 4) depthwise 3x3 (first C filters) -> slotB
  {
    long long tot = (long long)BB * CC * NPIX;
    dwconv3x3<<<(unsigned)((tot + 255) / 256), 256, 0, stream>>>(
        slotA, dw_w, dw_b, slotB, CC, tot);
  }
  // 5) qdw 1x1 -> qy in slotA
  conv1x1_gemm<_Float16, _Float16><<<dim3(NPIX / 64, CC / 64, BB), 128, 0,
                                     stream>>>(slotB, w_qdw16, qdw_b, slotA, CC,
                                               CC);
  // 6) inverse L2 norms of q rows and k rows
  rownorm_inv<<<BB * CC, 256, 0, stream>>>(slotA, rq, CC);
  rownorm_inv<<<BB * CC, 256, 0, stream>>>(t_kv, rk, 2 * CC);
  // 7) Gram partials (K split 8 ways, fp32 atomics), then fused softmax
  chan_gram<<<dim3(BB * NH, KSPLIT), 128, 0, stream>>>(slotA, t_kv, gbuf);
  attn_softmax<<<BB * NH, 32, 0, stream>>>(gbuf, rq, rk, temp, attnbuf);
  // 8) attn @ v -> slotB
  apply_attn<<<dim3(NPIX / 64, NH, BB), 128, 0, stream>>>(attnbuf, t_kv, slotB);
  // 9) proj 1x1 (+bias), fp32 output
  conv1x1_gemm<_Float16, float><<<dim3(NPIX / 64, CC / 64, BB), 128, 0,
                                  stream>>>(slotB, w_proj16, proj_b,
                                            (float*)d_out, CC, CC);
  (void)in_sizes; (void)n_in; (void)out_size; (void)ws_size;
}
